// ContrastiveLoss_38285338477146
// MI455X (gfx1250) — matrix operand order repaired and checked by reading, not compiled, not run on hardware
//
#include <hip/hip_runtime.h>

typedef __attribute__((ext_vector_type(16))) _Float16 v16h;
typedef __attribute__((ext_vector_type(8)))  _Float16 v8h;
typedef __attribute__((ext_vector_type(4)))  _Float16 v4h;
typedef __attribute__((ext_vector_type(8)))  float    v8f;

#define B_ 64
#define N_ 1024
#define D_ 128
#define RS_ 136                  // padded LDS row stride in halves (272B: 16B-aligned, bank-skewed)
#define MARGIN_SHIFT (-0.15f)    // sim - 0.5 + 0.35

// ---- async global->LDS path (CDNA5), with guaranteed-compile fallback ------
#if __has_builtin(__builtin_amdgcn_global_load_async_to_lds_b128)
#define USE_ASYNC_LDS 1
#if __has_builtin(__builtin_amdgcn_s_wait_asynccnt)
#define WAIT_ASYNC() __builtin_amdgcn_s_wait_asynccnt(0)
#else
#define WAIT_ASYNC() asm volatile("s_wait_asynccnt 0x0" ::: "memory")
#endif
// Builtin signature (from hipcc diagnostic): (int4 AS1*, int4 AS3*, Ii, Ii)
typedef int v4i_ __attribute__((vector_size(16)));
typedef __attribute__((address_space(1))) v4i_* g_v4i_p;
typedef __attribute__((address_space(3))) v4i_* l_v4i_p;
static __device__ __forceinline__ g_v4i_p to_g(const void* p) {
  return (g_v4i_p)(uintptr_t)p;            // flat address == global address
}
static __device__ __forceinline__ l_v4i_p to_l(void* p) {
  return (l_v4i_p)p;                       // addrspacecast flat -> local
}
#else
#define USE_ASYNC_LDS 0
#endif

// ---------------------------------------------------------------------------
// 1) L2-normalize each D=128 row, emit f16. One wave32 per row, 4 floats/lane.
// ---------------------------------------------------------------------------
__global__ __launch_bounds__(256)
void normalize_f16_kernel(const float* __restrict__ emb,
                          _Float16* __restrict__ out) {
  int tid  = blockIdx.x * blockDim.x + threadIdx.x;
  int row  = tid >> 5;
  int lane = tid & 31;
  const float4* src = (const float4*)(emb + (size_t)row * D_) + lane;
  float4 v = *src;
  float ss = v.x * v.x + v.y * v.y + v.z * v.z + v.w * v.w;
#pragma unroll
  for (int m = 16; m > 0; m >>= 1) ss += __shfl_xor(ss, m, 32);
  float s = 1.0f / fmaxf(sqrtf(ss), 1e-12f);
  v4h o;
  o.x = (_Float16)(v.x * s);
  o.y = (_Float16)(v.y * s);
  o.z = (_Float16)(v.z * s);
  o.w = (_Float16)(v.w * s);
  *(v4h*)(out + (size_t)row * D_ + lane * 4) = o;
}

// ---------------------------------------------------------------------------
// 2) Per-batch positive/negative counts. One block per batch.
// ---------------------------------------------------------------------------
__global__ __launch_bounds__(256)
void label_stats_kernel(const int* __restrict__ labels,
                        int* __restrict__ n_pos, int* __restrict__ n_neg) {
  __shared__ int sp, sn;
  int b = blockIdx.x;
  if (threadIdx.x == 0) { sp = 0; sn = 0; }
  __syncthreads();
  int lp = 0, ln = 0;
  for (int i = threadIdx.x; i < N_; i += blockDim.x) {
    int l = labels[b * N_ + i];
    lp += (l == 1);
    ln += (l == 0);
  }
#pragma unroll
  for (int m = 16; m > 0; m >>= 1) {
    lp += __shfl_xor(lp, m, 32);
    ln += __shfl_xor(ln, m, 32);
  }
  if ((threadIdx.x & 31) == 0) { atomicAdd(&sp, lp); atomicAdd(&sn, ln); }
  __syncthreads();
  if (threadIdx.x == 0) { n_pos[b] = sp; n_neg[b] = sn; }
}

// ---------------------------------------------------------------------------
// 3) coeff[b] = valid ? 1/(n_neg[b] * count) : 0 ; zero the output scalar.
// ---------------------------------------------------------------------------
__global__ void finalize_coeff_kernel(const int* __restrict__ n_pos,
                                      const int* __restrict__ n_neg,
                                      float* __restrict__ coeff,
                                      float* __restrict__ out) {
  if (threadIdx.x == 0 && blockIdx.x == 0) {
    long long cnt = 0;
    for (int b = 0; b < B_; ++b)
      if (n_pos[b] > 0 && n_neg[b] > 0) cnt += n_pos[b];
    float invc = 1.0f / (float)(cnt > 0 ? cnt : 1);
    for (int b = 0; b < B_; ++b) {
      bool valid = (n_pos[b] > 0) && (n_neg[b] > 0);
      float dn = (float)(n_neg[b] > 0 ? n_neg[b] : 1);
      coeff[b] = valid ? (invc / dn) : 0.0f;
    }
    out[0] = 0.0f;
  }
}

// ---------------------------------------------------------------------------
// 4) Masked-hinge Gram via V_WMMA_F32_16X16X32_F16 with LDS-staged,
//    double-buffered B panels (async global->LDS prefetch when available).
//    grid = (4 row-chunks, B). 8 waves/block; each wave owns 32 rows
//    (two 16-row A panels -> 8 WMMAs per staged B tile).
// ---------------------------------------------------------------------------
__global__ __launch_bounds__(256)
void hinge_gemm_wmma_kernel(const _Float16* __restrict__ E,
                            const int* __restrict__ labels,
                            const float* __restrict__ coeff,
                            float* __restrict__ out) {
  __shared__ _Float16 sB[2][16 * RS_];   // double-buffered 16x128 B panel
  __shared__ float    sNeg[N_];          // negative-column mask
  __shared__ float    ssum;

  const int b    = blockIdx.y;
  const int tid  = threadIdx.x;
  const int wave = tid >> 5;
  const int lane = tid & 31;
  const int grp  = lane >> 4;            // 0: lanes 0-15, 1: lanes 16-31
  const int l15  = lane & 15;
  const int rowBase = blockIdx.x * 256 + wave * 32;

  const _Float16* Eb = E + (size_t)b * N_ * D_;
  const int*      Lb = labels + b * N_;

  // Stage the negative mask once (uniform loop: no EXEC divergence).
  for (int i = tid; i < N_; i += 256) sNeg[i] = (Lb[i] == 0) ? 1.0f : 0.0f;
  if (tid == 0) ssum = 0.0f;

  // A fragments, ISA 16-bit 16x32 layout: two 16-row panels per wave.
  // lane l (m=l&15, g=l>>4): elems 0..7 -> K = kbase+8g+e ; 8..15 -> +16.
  v16h afrag[2][4];
#pragma unroll
  for (int p = 0; p < 2; ++p) {
    const _Float16* arow = Eb + (size_t)(rowBase + p * 16 + l15) * D_;
#pragma unroll
    for (int c = 0; c < 4; ++c) {
      const int k0 = c * 32 + 8 * grp;
      v8h lo = *(const v8h*)(arow + k0);
      v8h hi = *(const v8h*)(arow + k0 + 16);
      v16h a;
#pragma unroll
      for (int t = 0; t < 8; ++t) { a[t] = lo[t]; a[t + 8] = hi[t]; }
      afrag[p][c] = a;
    }
  }

  // Cooperative staging coords: thread -> (row 0..15, 16B chunk 0..15).
  const int srow   = tid >> 4;
  const int schunk = tid & 15;
  const _Float16* gstage = Eb + (size_t)srow * D_ + schunk * 8;
  const int loff = srow * RS_ + schunk * 8;

  // Prefetch tile 0 into buffer 0.
#if USE_ASYNC_LDS
  __builtin_amdgcn_global_load_async_to_lds_b128(
      to_g(gstage), to_l(&sB[0][loff]), 0, 0);
  WAIT_ASYNC();
#else
  *(uint4*)&sB[0][loff] = *(const uint4*)gstage;
#endif
  __syncthreads();   // covers sNeg + tile 0

  float acc[2][8];
#pragma unroll
  for (int p = 0; p < 2; ++p)
#pragma unroll
    for (int r = 0; r < 8; ++r) acc[p][r] = 0.0f;

  for (int jt = 0; jt < N_ / 16; ++jt) {
    const int cur = jt & 1;

    // Prefetch next tile into the other buffer (uniform scalar branch).
#if USE_ASYNC_LDS
    if (jt + 1 < N_ / 16)
      __builtin_amdgcn_global_load_async_to_lds_b128(
          to_g(gstage + (size_t)(jt + 1) * 16 * D_),
          to_l(&sB[1 - cur][loff]), 0, 0);
#else
    uint4 tnext;
    if (jt + 1 < N_ / 16)
      tnext = *(const uint4*)(gstage + (size_t)(jt + 1) * 16 * D_);
#endif

    // B fragments from LDS, ISA layout: elems 0..7 -> K = kbase+8g+e ; 8..15 -> +16.
    const _Float16* bb = &sB[cur][l15 * RS_];
    v16h bfrag[4];
#pragma unroll
    for (int c = 0; c < 4; ++c) {
      const int k0 = c * 32 + 8 * grp;
      v8h lo = *(const v8h*)(bb + k0);
      v8h hi = *(const v8h*)(bb + k0 + 16);
      v16h x;
#pragma unroll
      for (int t = 0; t < 8; ++t) { x[t] = lo[t]; x[t + 8] = hi[t]; }
      bfrag[c] = x;
    }

    const float nf = sNeg[jt * 16 + l15];

    v8f c0 = {}, c1 = {};
#pragma unroll
    for (int c = 0; c < 4; ++c)
      c0 = __builtin_amdgcn_wmma_f32_16x16x32_f16(
          false, afrag[0][c], false, bfrag[c], (short)0, c0, false, false);
#pragma unroll
    for (int c = 0; c < 4; ++c)
      c1 = __builtin_amdgcn_wmma_f32_16x16x32_f16(
          false, afrag[1][c], false, bfrag[c], (short)0, c1, false, false);

    // C/D layout: VGPR r -> M = r (lanes 0-15) / M = r+8 (lanes 16-31)
#pragma unroll
    for (int r = 0; r < 8; ++r) {
      float h0 = c0[r] + MARGIN_SHIFT;
      float h1 = c1[r] + MARGIN_SHIFT;
      h0 = h0 > 0.0f ? h0 : 0.0f;
      h1 = h1 > 0.0f ? h1 : 0.0f;
      acc[0][r] += nf * h0;
      acc[1][r] += nf * h1;
    }

#if USE_ASYNC_LDS
    WAIT_ASYNC();        // next buffer resident
    __syncthreads();     // everyone done reading cur, next buffer visible
#else
    __syncthreads();     // everyone done reading cur
    if (jt + 1 < N_ / 16) *(uint4*)&sB[1 - cur][loff] = tnext;
    __syncthreads();
#endif
  }

  // Reduce each row over the 16 lanes of its group (xor<16 keeps bit4).
#pragma unroll
  for (int p = 0; p < 2; ++p)
#pragma unroll
    for (int r = 0; r < 8; ++r)
#pragma unroll
      for (int m = 1; m < 16; m <<= 1)
        acc[p][r] += __shfl_xor(acc[p][r], m, 32);

  if (l15 == 0) {        // lanes 0 and 16 carry row totals
    float wsum = 0.0f;
#pragma unroll
    for (int p = 0; p < 2; ++p)
#pragma unroll
      for (int r = 0; r < 8; ++r) {
        const int i = rowBase + p * 16 + r + 8 * grp;
        if (Lb[i] == 1) wsum += acc[p][r];
      }
    atomicAdd(&ssum, wsum);
  }
  __syncthreads();
  if (tid == 0) atomicAdd(out, ssum * coeff[b]);
}

// ---------------------------------------------------------------------------
extern "C" void kernel_launch(void* const* d_in, const int* in_sizes, int n_in,
                              void* d_out, int out_size, void* d_ws, size_t ws_size,
                              hipStream_t stream) {
  const float* emb    = (const float*)d_in[0];
  const int*   labels = (const int*)d_in[1];
  float*       out    = (float*)d_out;

  char* ws = (char*)d_ws;
  _Float16* Eh = (_Float16*)ws;                         // 16 MB normalized f16
  size_t ehBytes = (size_t)B_ * N_ * D_ * sizeof(_Float16);
  int*   n_pos = (int*)(ws + ehBytes);
  int*   n_neg = n_pos + B_;
  float* coeff = (float*)(n_neg + B_);

  normalize_f16_kernel<<<(B_ * N_) / 8, 256, 0, stream>>>(emb, Eh);
  label_stats_kernel<<<B_, 256, 0, stream>>>(labels, n_pos, n_neg);
  finalize_coeff_kernel<<<1, 64, 0, stream>>>(n_pos, n_neg, coeff, out);

  dim3 grid(4, B_);
  hinge_gemm_wmma_kernel<<<grid, 256, 0, stream>>>(Eh, labels, coeff, out);
}
